// BiRNN_35708358099418
// MI455X (gfx1250) — matrix operand order repaired and checked by reading, not compiled, not run on hardware
//
#include <hip/hip_runtime.h>

// ---------------- problem constants ----------------
constexpr int kT  = 2048;    // seq len
constexpr int kB  = 32;      // batch
constexpr int kIN = 256;     // layer-0 input dim
constexpr int kH  = 256;     // hidden
constexpr int kG  = 4 * kH;  // 1024 gate columns (i,f,g,o)
constexpr int kM  = kT * kB; // 65536 rows for input projection

// ---------------- types ----------------
typedef __bf16 bf16;
typedef __attribute__((ext_vector_type(16))) __bf16 v16bf;
typedef __attribute__((ext_vector_type(8)))  __bf16 bf16x8;
typedef __attribute__((ext_vector_type(8)))  float  v8f;

static __device__ inline v16bf combine16(bf16x8 a, bf16x8 b) {
  return __builtin_shufflevector(a, b, 0,1,2,3,4,5,6,7,8,9,10,11,12,13,14,15);
}
static __device__ inline float sigmoid_(float x) { return 1.0f / (1.0f + __expf(-x)); }
static __device__ inline float tanh_(float x) {
  float e = __expf(-2.0f * x);
  return (1.0f - e) / (1.0f + e);
}

// LDS byte offset of a generic pointer to __shared__ (low 32 bits = DS address)
static __device__ inline unsigned lds_off(const void* p) { return (unsigned)(size_t)p; }

// CDNA5 async global->LDS copy, 16B per lane, tracked by ASYNCcnt.
static __device__ inline void async_g2l_b128(unsigned lds_byte_off, const void* gaddr) {
  asm volatile("global_load_async_to_lds_b128 %0, %1, off"
               :: "v"(lds_byte_off), "v"(gaddr) : "memory");
}
static __device__ inline void wait_async0() {
#if __has_builtin(__builtin_amdgcn_s_wait_asynccnt)
  __builtin_amdgcn_s_wait_asynccnt(0);
#else
  asm volatile("s_wait_asynccnt 0x0" ::: "memory");
#endif
}

// ---------------- elementwise converters ----------------
__global__ void cvt_f32_bf16(const float* __restrict__ in, bf16* __restrict__ out, int n) {
  int i = blockIdx.x * blockDim.x + threadIdx.x;
  if (i < n) out[i] = (bf16)in[i];
}
__global__ void bias_sum(const float* __restrict__ a, const float* __restrict__ b,
                         float* __restrict__ out, int n) {
  int i = blockIdx.x * blockDim.x + threadIdx.x;
  if (i < n) out[i] = a[i] + b[i];
}

// ---------------- input projection: xg = X @ W^T + bias ----------------
__global__ __launch_bounds__(256) void gemm_xg(
    const bf16* __restrict__ X, const bf16* __restrict__ Wih,
    const float* __restrict__ bias, float* __restrict__ xg, int D) {
  const int dir = blockIdx.y;
  const bf16*  W   = Wih  + (size_t)dir * kG * D;
  const float* bs  = bias + dir * kG;
  float*       out = xg   + (size_t)dir * kM * kG;

  const int lane = threadIdx.x & 31;
  const int l15  = lane & 15;
  const int half = lane >> 4;
  const int wid  = blockIdx.x * 8 + (threadIdx.x >> 5);
  const int mtile  = wid >> 4;
  const int ngroup = wid & 15;
  const size_t mrow = (size_t)(mtile * 16 + l15);

  v8f acc[4] = {};
  const int KT = D >> 5;
  for (int kt = 0; kt < KT; ++kt) {
    const bf16* arow = X + mrow * D + kt * 32;
    v16bf A = combine16(*(const bf16x8*)(arow + half * 8),
                        *(const bf16x8*)(arow + 16 + half * 8));
#pragma unroll
    for (int ntl = 0; ntl < 4; ++ntl) {
      int n = ngroup * 64 + ntl * 16 + l15;
      const bf16x8* pb = (const bf16x8*)(W + (size_t)n * D + kt * 32);
      v16bf Bf = combine16(pb[half], pb[2 + half]);
      acc[ntl] = __builtin_amdgcn_wmma_f32_16x16x32_bf16(
          false, A, false, Bf, (short)0, acc[ntl], false, false);
    }
    if (kt + 1 < KT) __builtin_prefetch(X + mrow * D + (kt + 1) * 32, 0, 3);
  }
#pragma unroll
  for (int ntl = 0; ntl < 4; ++ntl) {
    int n = ngroup * 64 + ntl * 16 + l15;
    float bv = bs[n];
#pragma unroll
    for (int r = 0; r < 8; ++r) {
      int mm = mtile * 16 + r + 8 * half;
      out[(size_t)mm * kG + n] = acc[ntl][r] + bv;
    }
  }
}

// ---------------- persistent bidirectional recurrence ----------------
// grid = 2 (dir), block = 512 (16 waves). Wave ut owns hidden-unit tile
// [ut*16, ut*16+16) and ALL FOUR of its gate N-tiles -> whole cell update in
// registers (c never leaves VGPRs). w_hh resident as WMMA B-fragments (256
// VGPRs/wave). xg double-buffered in LDS via async global->LDS copies; h
// double-buffered in LDS; ONE barrier per timestep.
constexpr int kXROW = kG + 4;                       // padded LDS row (floats)
constexpr unsigned kXBUF = kB * kXROW * 4;          // 131584 B per xg buffer
constexpr unsigned kHBUF = kB * kH * 2;             // 16384 B per h buffer

// copy xg[ts] (kB x kG f32, rows 4 KB contiguous) into padded LDS buffer
static __device__ inline void async_copy_xg(const float* __restrict__ gsrc,
                                            unsigned ldsbase, int tid) {
  const int row = tid >> 4;   // 0..31
  const int sub = tid & 15;   // 16 threads per row, 256 B each
  const char* g = (const char*)gsrc + row * 4096 + sub * 256;
  unsigned    l = ldsbase + row * (kXROW * 4) + sub * 256;
#pragma unroll
  for (int j = 0; j < 16; ++j)
    async_g2l_b128(l + j * 16, g + j * 16);
}

__global__ __launch_bounds__(512) void lstm_recurrent(
    const float* __restrict__ xg,     // [2][kT][kB][kG]
    const bf16*  __restrict__ whh,    // [2][kG][kH]
    float*       __restrict__ out)    // [kT][kB][2*kH]
{
  extern __shared__ char smem[];
  // layout: [xg buf0][xg buf1][h buf0][h buf1] -- pointers computed per use
  // (no pointer arrays: aggregate initializers with addrspacecast of the
  //  shared-memory symbol become unsupported static initializers)

  const int dir  = blockIdx.x;
  const float* xgd = xg  + (size_t)dir * kT * kB * kG;
  const bf16*  W   = whh + (size_t)dir * kG * kH;
  const int tid  = threadIdx.x;
  const int ut   = tid >> 5;          // wave 0..15 = hidden-unit tile
  const int lane = tid & 31;
  const int l15  = lane & 15;
  const int half = lane >> 4;
  const int u    = ut * 16 + l15;     // this lane's hidden unit column

  {
    bf16* h0 = (bf16*)(smem + 2 * kXBUF);
    for (int i = tid; i < kB * kH; i += 512) h0[i] = (bf16)0.0f;
  }

  // resident w_hh fragments: 4 gates x 8 K-tiles (B frag: lane=col n, k = 8*half+s, then 16+8*half+(s-8))
  v16bf Bf[4][8];
#pragma unroll
  for (int g = 0; g < 4; ++g) {
    int n = g * kH + u;               // gate column in [0,1024)
#pragma unroll
    for (int kt = 0; kt < 8; ++kt) {
      const bf16x8* pb = (const bf16x8*)(W + (size_t)n * kH + kt * 32);
      Bf[g][kt] = combine16(pb[half], pb[2 + half]);
    }
  }

  v8f c[2] = {};                      // cell state, register-resident

  // prologue: fill xg buffer 0 for first timestep
  const int ts0 = dir ? (kT - 1) : 0;
  async_copy_xg(xgd + (size_t)ts0 * kB * kG, lds_off(smem), tid);
  wait_async0();
  __syncthreads();

  for (int t = 0; t < kT; ++t) {
    const int ts  = dir ? (kT - 1 - t) : t;
    const unsigned cur = t & 1, nxt = cur ^ 1;

    // prefetch next timestep's xg while we do WMMA
    if (t + 1 < kT) {
      const int tsn = dir ? (kT - 2 - t) : (t + 1);
      async_copy_xg(xgd + (size_t)tsn * kB * kG, lds_off(smem) + nxt * kXBUF, tid);
    }

    // gates = h @ w_hh^T : 64 WMMAs/wave, A from LDS h buffer
    v8f acc[4][2] = {};
    const bf16* hc = (const bf16*)(smem + 2 * kXBUF + cur * kHBUF);
#pragma unroll
    for (int kt = 0; kt < 8; ++kt) {
      v16bf A[2];
#pragma unroll
      for (int mt = 0; mt < 2; ++mt) {
        const bf16* arow = hc + (mt * 16 + l15) * kH + kt * 32;
        A[mt] = combine16(*(const bf16x8*)(arow + half * 8),
                          *(const bf16x8*)(arow + 16 + half * 8));
      }
#pragma unroll
      for (int g = 0; g < 4; ++g)
#pragma unroll
        for (int mt = 0; mt < 2; ++mt)
          acc[g][mt] = __builtin_amdgcn_wmma_f32_16x16x32_bf16(
              false, A[mt], false, Bf[g][kt], (short)0, acc[g][mt], false, false);
    }

    // fused cell update, fully in registers (i,f,g,o + c share lane mapping)
    const float* xr = (const float*)(smem + cur * kXBUF);
    float* outt = out + (size_t)ts * kB * (2 * kH) + dir * kH;
    bf16*  hn   = (bf16*)(smem + 2 * kXBUF + nxt * kHBUF);
#pragma unroll
    for (int mt = 0; mt < 2; ++mt) {
#pragma unroll
      for (int r = 0; r < 8; ++r) {
        const int b = mt * 16 + r + 8 * half;
        const float gi = acc[0][mt][r] + xr[b * kXROW + u];
        const float gf = acc[1][mt][r] + xr[b * kXROW + kH + u];
        const float gg = acc[2][mt][r] + xr[b * kXROW + 2 * kH + u];
        const float go = acc[3][mt][r] + xr[b * kXROW + 3 * kH + u];
        const float ci = sigmoid_(gf) * c[mt][r] + sigmoid_(gi) * tanh_(gg);
        const float h  = sigmoid_(go) * tanh_(ci);
        c[mt][r] = ci;
        hn[b * kH + u] = (bf16)h;
        outt[b * (2 * kH) + u] = h;
      }
    }

    wait_async0();     // our async copies for buf[nxt] are done
    __syncthreads();   // everyone's copies + h writes visible
  }
}

// ---------------- host launcher ----------------
extern "C" void kernel_launch(void* const* d_in, const int* in_sizes, int n_in,
                              void* d_out, int out_size, void* d_ws, size_t ws_size,
                              hipStream_t stream) {
  const float* x    = (const float*)d_in[0];
  const float* wih0 = (const float*)d_in[1];
  const float* whh0 = (const float*)d_in[2];
  const float* bih0 = (const float*)d_in[3];
  const float* bhh0 = (const float*)d_in[4];
  const float* wih1 = (const float*)d_in[5];
  const float* whh1 = (const float*)d_in[6];
  const float* bih1 = (const float*)d_in[7];
  const float* bhh1 = (const float*)d_in[8];

  char* ws = (char*)d_ws;
  bf16*  xbf   = (bf16*) (ws);                      //  64 MB
  bf16*  wihbf = (bf16*) (ws + (64ull  << 20));     //   2 MB
  bf16*  whhbf = (bf16*) (ws + (66ull  << 20));     //   1 MB
  float* bias  = (float*)(ws + (67ull  << 20));     //   8 KB
  float* xg    = (float*)(ws + (68ull  << 20));     // 512 MB
  float* out0  = (float*)(ws + (580ull << 20));     // 128 MB

  const size_t smem = 2 * (size_t)kXBUF + 2 * (size_t)kHBUF;   // ~289 KB
  auto cvt = [&](const float* src, bf16* dst, int n) {
    cvt_f32_bf16<<<(n + 255) / 256, 256, 0, stream>>>(src, dst, n);
  };

  // ---- layer 0 (D = 256) ----
  cvt(x, xbf, kM * kIN);
  cvt(wih0, wihbf, 2 * kG * kIN);
  cvt(whh0, whhbf, 2 * kG * kH);
  bias_sum<<<8, 256, 0, stream>>>(bih0, bhh0, bias, 2 * kG);
  gemm_xg<<<dim3(kM / 8, 2), 256, 0, stream>>>(xbf, wihbf, bias, xg, kIN);
  lstm_recurrent<<<2, 512, smem, stream>>>(xg, whhbf, out0);

  // ---- layer 1 (D = 512) ----
  cvt(out0, xbf, kM * 2 * kH);
  cvt(wih1, wihbf, 2 * kG * 2 * kH);
  cvt(whh1, whhbf, 2 * kG * kH);
  bias_sum<<<8, 256, 0, stream>>>(bih1, bhh1, bias, 2 * kG);
  gemm_xg<<<dim3(kM / 8, 2), 256, 0, stream>>>(xbf, wihbf, bias, xg, 2 * kH);
  lstm_recurrent<<<2, 512, smem, stream>>>(xg, whhbf, (float*)d_out);
}
